// CTDAutoEncoder_81990925680756
// MI455X (gfx1250) — matrix operand order repaired and checked by reading, not compiled, not run on hardware
//
#include <hip/hip_runtime.h>
#include <math.h>

// ---------------------------------------------------------------------------
// CDNA5 (gfx1250, wave32) WMMA implementation of the CTD autoencoder forward.
// All matmul-like work (convs via implicit GEMM, transformer GEMMs, attention,
// VQ distance) runs on v_wmma_f32_16x16x32_f16. fp32 params are repacked to
// f16 [N][K] row-major so B-fragments are contiguous 32B loads per lane.
// Accumulators are indexed only by compile-time constants (templated tile
// counts) so they stay in fixed VGPRs. Conv kernels are templated on
// <KH, TRANSPOSED> so tap math folds; OOB reads go through a zero page
// (branch-free, EXEC stays all ones). GEMM K-loop is software pipelined and
// tiled 32(M) x 64(N) per wave (8 WMMA per 6 fragment loads).
// ---------------------------------------------------------------------------

typedef __attribute__((ext_vector_type(16))) _Float16 v16h;
typedef __attribute__((ext_vector_type(8)))  _Float16 v8h;
typedef __attribute__((ext_vector_type(8)))  float    v8f;
typedef _Float16 half_t;

#define CDIM   256
#define NTOK   4096          // B * NCODES
#define KCODE  1024

// ---- WMMA fragment helpers ------------------------------------------------
// A fragment in the contiguous k order used here: lane l holds row m=l&15; lanes<16 hold K {0..15},
// lanes>=16 hold K {16..31}. caller passes rowp = rowbase + (l>>4)*16.
__device__ __forceinline__ v16h frag_a(const half_t* rowp) {
  v8h lo = *(const v8h*)rowp;
  v8h hi = *(const v8h*)(rowp + 8);
  v16h a;
#pragma unroll
  for (int i = 0; i < 8; ++i) { a[i] = lo[i]; a[i + 8] = hi[i]; }
  return a;
}
// B fragment, same contiguous k order as frag_a: lane l holds column n=l&15, K range (l>>4)*16..+15,
// contiguous when weights are stored [N][K] row-major.
__device__ __forceinline__ v16h frag_b(const half_t* p) {
  return *(const v16h*)p;
}
__device__ __forceinline__ v8f wmma_step(v16h a, v16h b, v8f c) {
  return __builtin_amdgcn_wmma_f32_16x16x32_f16(false, a, false, b, (short)0, c,
                                                false, false);
}

// ---- generic GEMM: Y[M,N] = epilogue(A[M,K] @ Bw[N,K]^T) --------------------
// one wave per block; wave computes a MT*16(M) x NT*16(N) tile. K-loop is
// software pipelined (fragments for step k+1 loaded before step k's WMMAs).
template <int MT, int NT>
__global__ void gemm_wmma(const half_t* __restrict__ A, int lda,
                          const half_t* __restrict__ Bw, int K,
                          half_t* __restrict__ Yh, float* __restrict__ Yf,
                          int ldy, const float* __restrict__ bias,
                          const half_t* __restrict__ resid, int relu) {
  const int lane = threadIdx.x & 31;
  const int row0 = blockIdx.x * (MT * 16);
  const int nb   = blockIdx.y * (NT * 16);

  const half_t* aptr[MT];
#pragma unroll
  for (int mi = 0; mi < MT; ++mi)
    aptr[mi] = A + (size_t)(row0 + mi * 16 + (lane & 15)) * lda +
               ((lane >> 4) * 16);

  const half_t* bp[NT];
#pragma unroll
  for (int j = 0; j < NT; ++j) {
    const int ncol = nb + j * 16 + (lane & 15);
    bp[j] = Bw + (size_t)ncol * K + ((lane >> 4) * 16);
  }

  v8f acc[MT][NT];
  v8f zf = {};
#pragma unroll
  for (int mi = 0; mi < MT; ++mi)
#pragma unroll
    for (int j = 0; j < NT; ++j) acc[mi][j] = zf;

  v16h a[MT];
#pragma unroll
  for (int mi = 0; mi < MT; ++mi) a[mi] = frag_a(aptr[mi]);
  v16h bc[NT];
#pragma unroll
  for (int j = 0; j < NT; ++j) bc[j] = frag_b(bp[j]);

  for (int k0 = 32; k0 < K; k0 += 32) {
    v16h an[MT];
#pragma unroll
    for (int mi = 0; mi < MT; ++mi) an[mi] = frag_a(aptr[mi] + k0);
    v16h bn[NT];
#pragma unroll
    for (int j = 0; j < NT; ++j) bn[j] = frag_b(bp[j] + k0);
#pragma unroll
    for (int mi = 0; mi < MT; ++mi)
#pragma unroll
      for (int j = 0; j < NT; ++j)
        acc[mi][j] = wmma_step(a[mi], bc[j], acc[mi][j]);
#pragma unroll
    for (int mi = 0; mi < MT; ++mi) a[mi] = an[mi];
#pragma unroll
    for (int j = 0; j < NT; ++j) bc[j] = bn[j];
  }
#pragma unroll
  for (int mi = 0; mi < MT; ++mi)
#pragma unroll
    for (int j = 0; j < NT; ++j)
      acc[mi][j] = wmma_step(a[mi], bc[j], acc[mi][j]);

#pragma unroll
  for (int mi = 0; mi < MT; ++mi)
#pragma unroll
    for (int j = 0; j < NT; ++j) {
      const int col = nb + j * 16 + (lane & 15);
#pragma unroll
      for (int i = 0; i < 8; ++i) {
        const int row = row0 + mi * 16 + ((lane >> 4) * 8) + i;
        float v = acc[mi][j][i];
        if (bias)  v += bias[col];
        if (relu)  v = v > 0.f ? v : 0.f;
        if (resid) v += (float)resid[(size_t)row * ldy + col];
        if (Yh) Yh[(size_t)row * ldy + col] = (half_t)v;
        else    Yf[(size_t)row * ldy + col] = v;
      }
    }
}

// ---- conv / transposed-conv as implicit GEMM over taps ---------------------
// X: [NB, IH, IW, C] f16 channel-last.  Wk: [C_out][taps*C] f16 (K order =
// tap*C + c_in).  OOB lanes read a zero page -> branch-free inner loop.
// Epilogue: y = relu?(acc*alpha[c]+beta[c]) (+resid).
template <int KH, int TR>
__global__ void convlike_wmma(const half_t* __restrict__ X,
                              const half_t* __restrict__ Wk,
                              const float* __restrict__ alpha,
                              const float* __restrict__ beta,
                              const half_t* __restrict__ resid,
                              half_t* __restrict__ Y,
                              const half_t* __restrict__ zpad,
                              int IH, int lw, int stride, int pad, int relu) {
  const int OW = 1 << lw;
  const int lane = threadIdx.x & 31;
  const int row0 = blockIdx.x * 16;
  const int nb   = blockIdx.y * 64;
  const int m  = row0 + (lane & 15);
  const int ox = m & (OW - 1);
  const int oy = (m >> lw) & (OW - 1);
  const int bb = m >> (2 * lw);

  constexpr int TAPS = KH * KH;
  const half_t* wp[4];
#pragma unroll
  for (int j = 0; j < 4; ++j) {
    const int ncol = nb + j * 16 + (lane & 15);
    wp[j] = Wk + (size_t)ncol * (TAPS * CDIM) + ((lane >> 4) * 16);
  }
  const half_t* zp = zpad + ((lane >> 4) * 16);

  v8f acc[4];
  v8f zf = {};
#pragma unroll
  for (int j = 0; j < 4; ++j) acc[j] = zf;

  for (int tt = 0; tt < TAPS; ++tt) {
    const int ky = tt / KH, kx = tt % KH;   // KH is constexpr -> folds
    int iy, ix; bool ok;
    if (TR) {                    // stride-2 ConvTranspose gather form
      const int ry = oy + pad - ky;
      const int rx = ox + pad - kx;
      iy = ry >> 1; ix = rx >> 1;
      ok = (ry >= 0) && ((ry & 1) == 0) && (iy < IH) &&
           (rx >= 0) && ((rx & 1) == 0) && (ix < IH);
    } else {
      iy = oy * stride - pad + ky;
      ix = ox * stride - pad + kx;
      ok = (iy >= 0) && (iy < IH) && (ix >= 0) && (ix < IH);
    }
    const half_t* rowp = X + (((size_t)bb * IH + iy) * IH + ix) * CDIM +
                         ((lane >> 4) * 16);
    const half_t* src = ok ? rowp : zp;     // branch-free OOB -> zero page
    const int woff = tt * CDIM;
#pragma unroll
    for (int k0 = 0; k0 < CDIM; k0 += 32) {
      v16h a = frag_a(src + k0);
#pragma unroll
      for (int j = 0; j < 4; ++j)
        acc[j] = wmma_step(a, frag_b(wp[j] + woff + k0), acc[j]);
    }
  }
#pragma unroll
  for (int j = 0; j < 4; ++j) {
    const int col = nb + j * 16 + (lane & 15);
    const float al = alpha[col], be = beta[col];
#pragma unroll
    for (int i = 0; i < 8; ++i) {
      const int row = row0 + ((lane >> 4) * 8) + i;
      float v = acc[j][i] * al + be;
      if (relu)  v = v > 0.f ? v : 0.f;
      if (resid) v += (float)resid[(size_t)row * CDIM + col];
      Y[(size_t)row * CDIM + col] = (half_t)v;
    }
  }
}

// ---- attention: one workgroup per (batch, head); seq=64, hd=64 -------------
__global__ void attention_wmma(const half_t* __restrict__ qkv,
                               half_t* __restrict__ O) {
  __shared__ __align__(32) float  sS[64][64];
  __shared__ __align__(32) half_t sP[64][80];   // pitch 80 -> 160B rows
  __shared__ __align__(32) half_t sV[64][80];   // V^T: [d][key]
  const int tid  = threadIdx.x;      // 0..127 (4 waves)
  const int lane = tid & 31;
  const int wv   = tid >> 5;
  const int b = blockIdx.x >> 2;
  const int h = blockIdx.x & 3;
  const int tokbase = b * 64;
  const int qoff = h * 64, koff = 256 + h * 64, voff = 512 + h * 64;

  for (int e = tid; e < 4096; e += 128) {
    const int key = e >> 6, d = e & 63;
    sV[d][key] = qkv[(size_t)(tokbase + key) * 768 + voff + d];
  }
  {   // S = Q K^T, rows 16*wv..+15
    const int arow = tokbase + wv * 16 + (lane & 15);
    const half_t* aptr = qkv + (size_t)arow * 768 + qoff + ((lane >> 4) * 16);
    const half_t* kb[4];
#pragma unroll
    for (int j = 0; j < 4; ++j) {
      const int n = j * 16 + (lane & 15);
      kb[j] = qkv + (size_t)(tokbase + n) * 768 + koff + ((lane >> 4) * 16);
    }
    v8f acc[4]; v8f zf = {};
#pragma unroll
    for (int j = 0; j < 4; ++j) acc[j] = zf;
#pragma unroll
    for (int k0 = 0; k0 < 64; k0 += 32) {
      v16h a = frag_a(aptr + k0);
#pragma unroll
      for (int j = 0; j < 4; ++j)
        acc[j] = wmma_step(a, frag_b(kb[j] + k0), acc[j]);
    }
#pragma unroll
    for (int j = 0; j < 4; ++j)
#pragma unroll
      for (int i = 0; i < 8; ++i)
        sS[wv * 16 + ((lane >> 4) * 8) + i][j * 16 + (lane & 15)] = acc[j][i];
  }
  __syncthreads();
  if (tid < 64) {   // softmax row tid (scale = hd^-0.5 = 1/8)
    float mx = -1e30f;
    for (int n = 0; n < 64; ++n) {
      const float s = sS[tid][n] * 0.125f;
      sS[tid][n] = s; mx = s > mx ? s : mx;
    }
    float sum = 0.f;
    for (int n = 0; n < 64; ++n) {
      const float e = __expf(sS[tid][n] - mx);
      sS[tid][n] = e; sum += e;
    }
    const float inv = 1.f / sum;
    for (int n = 0; n < 64; ++n) sP[tid][n] = (half_t)(sS[tid][n] * inv);
  }
  __syncthreads();
  {   // O = P V  (A from sP, B from sV=V^T)
    const int mrow = wv * 16 + (lane & 15);
    const half_t* aptr = &sP[mrow][(lane >> 4) * 16];
    const half_t* vb[4];
#pragma unroll
    for (int j = 0; j < 4; ++j)
      vb[j] = &sV[j * 16 + (lane & 15)][(lane >> 4) * 16];
    v8f acc[4]; v8f zf = {};
#pragma unroll
    for (int j = 0; j < 4; ++j) acc[j] = zf;
#pragma unroll
    for (int k0 = 0; k0 < 64; k0 += 32) {
      v16h a = frag_a(aptr + k0);
#pragma unroll
      for (int j = 0; j < 4; ++j)
        acc[j] = wmma_step(a, frag_b(vb[j] + k0), acc[j]);
    }
#pragma unroll
    for (int j = 0; j < 4; ++j)
#pragma unroll
      for (int i = 0; i < 8; ++i) {
        const int row = tokbase + wv * 16 + ((lane >> 4) * 8) + i;
        const int col = h * 64 + j * 16 + (lane & 15);
        O[(size_t)row * 256 + col] = (half_t)acc[j][i];
      }
  }
}

// ---- LayerNorm over C=256 (one wave per row), optional bias and pos-add ----
__global__ void ln_rows(const half_t* __restrict__ X, const float* __restrict__ g,
                        const float* __restrict__ b, const float* __restrict__ pos,
                        half_t* __restrict__ Y) {
  const int row = blockIdx.x;
  const int lane = threadIdx.x;
  const half_t* xp = X + (size_t)row * 256 + lane * 8;
  v8h xv = *(const v8h*)xp;
  float v[8], s = 0.f, ss = 0.f;
#pragma unroll
  for (int i = 0; i < 8; ++i) { v[i] = (float)xv[i]; s += v[i]; ss += v[i] * v[i]; }
#pragma unroll
  for (int o = 16; o > 0; o >>= 1) { s += __shfl_xor(s, o); ss += __shfl_xor(ss, o); }
  const float mean = s * (1.f / 256.f);
  const float var  = ss * (1.f / 256.f) - mean * mean;
  const float inv  = rsqrtf(var + 1e-5f);
  const int tok = row & 63;
  half_t* yp = Y + (size_t)row * 256 + lane * 8;
#pragma unroll
  for (int i = 0; i < 8; ++i) {
    const int c = lane * 8 + i;
    float hh = (v[i] - mean) * inv * g[c];
    if (b)   hh += b[c];
    if (pos) hh += pos[tok * 256 + c];
    yp[i] = (half_t)hh;
  }
}

// ---- small elementwise / prep kernels --------------------------------------
__global__ void embed_kernel(const int* __restrict__ tok,
                             const float* __restrict__ emb,
                             half_t* __restrict__ Y, int total) {
  const int i = blockIdx.x * blockDim.x + threadIdx.x;
  if (i >= total) return;
  Y[i] = (half_t)emb[tok[i >> 8] * 256 + (i & 255)];
}
__global__ void cvt_h_kernel(const float* __restrict__ s, half_t* __restrict__ d,
                             int n) {
  const int i = blockIdx.x * blockDim.x + threadIdx.x;
  if (i < n) d[i] = (half_t)s[i];
}
__global__ void zero_h_kernel(half_t* __restrict__ p, int n) {
  const int i = blockIdx.x * blockDim.x + threadIdx.x;
  if (i < n) p[i] = (half_t)0.f;
}
// conv weight repack -> [O][tap*I + ci] f16; transposed=1 for ConvTranspose
// (torch stores [I][O][KH][KW]).
__global__ void repack_conv_kernel(const float* __restrict__ w,
                                   half_t* __restrict__ out,
                                   int O, int I, int KH, int transposed,
                                   int total) {
  const int i = blockIdx.x * blockDim.x + threadIdx.x;
  if (i >= total) return;
  const int ci = i % I;
  const int t2 = i / I;
  const int tt = t2 % (KH * KH);
  const int o  = t2 / (KH * KH);
  const int ky = tt / KH, kx = tt % KH;
  const size_t src = transposed
      ? ((((size_t)ci * O + o) * KH + ky) * KH + kx)
      : ((((size_t)o * I + ci) * KH + ky) * KH + kx);
  out[i] = (half_t)w[src];
}
__global__ void alphabeta_kernel(const float* __restrict__ bias,
                                 const float* __restrict__ g,
                                 const float* __restrict__ be,
                                 float* __restrict__ alpha,
                                 float* __restrict__ beta, int n) {
  const int i = blockIdx.x * blockDim.x + threadIdx.x;
  if (i >= n) return;
  const float a = g[i] * rsqrtf(1.f + 1e-5f);   // BN eval: g / sqrt(1+eps)
  alpha[i] = a;
  beta[i]  = bias[i] * a + be[i];
}
__global__ void rownorm_kernel(const float* __restrict__ cb,
                               float* __restrict__ out) {
  const int row = blockIdx.x, lane = threadIdx.x;
  float s = 0.f;
  for (int c = lane; c < 256; c += 32) {
    const float v = cb[(size_t)row * 256 + c];
    s += v * v;
  }
#pragma unroll
  for (int o = 16; o > 0; o >>= 1) s += __shfl_xor(s, o);
  if (lane == 0) out[row] = s;
}
__global__ void vq_argmin_kernel(const float* __restrict__ D,
                                 const float* __restrict__ cbn,
                                 int* __restrict__ idx) {
  __shared__ float sv[256];
  __shared__ int   si[256];
  const int row = blockIdx.x, tid = threadIdx.x;
  float best = 1e30f; int bi = 0;
  for (int k = tid; k < KCODE; k += 256) {
    const float sc = cbn[k] - 2.f * D[(size_t)row * KCODE + k];
    if (sc < best) { best = sc; bi = k; }
  }
  sv[tid] = best; si[tid] = bi;
  __syncthreads();
  for (int s = 128; s > 0; s >>= 1) {
    if (tid < s) {
      if (sv[tid + s] < sv[tid] ||
          (sv[tid + s] == sv[tid] && si[tid + s] < si[tid])) {
        sv[tid] = sv[tid + s]; si[tid] = si[tid + s];
      }
    }
    __syncthreads();
  }
  if (tid == 0) idx[row] = si[0];
}
__global__ void vq_gather_kernel(const half_t* __restrict__ cbh,
                                 const int* __restrict__ idx,
                                 half_t* __restrict__ zq, int total) {
  const int i = blockIdx.x * blockDim.x + threadIdx.x;
  if (i >= total) return;
  zq[i] = cbh[(size_t)idx[i >> 8] * 256 + (i & 255)];
}

// ---------------------------------------------------------------------------
extern "C" void kernel_launch(void* const* d_in, const int* in_sizes, int n_in,
                              void* d_out, int out_size, void* d_ws,
                              size_t ws_size, hipStream_t stream) {
  (void)in_sizes; (void)n_in; (void)out_size; (void)ws_size;
  // ---- input map (JAX pytree order: sorted dict keys, lists in order) ------
  int p = 0;
  struct CBp { const float *b1,*b2,*be1,*be2,*bes,*bs,*g1,*g2,*gs,*w1,*w2,*ws; };
  struct TLp { const float *ln1,*ln2,*w1,*w2,*wo,*wqkv; };
  auto nf = [&]() -> const float* { return (const float*)d_in[p++]; };
  auto readCB = [&](CBp& c) { c.b1=nf(); c.b2=nf(); c.be1=nf(); c.be2=nf();
    c.bes=nf(); c.bs=nf(); c.g1=nf(); c.g2=nf(); c.gs=nf();
    c.w1=nf(); c.w2=nf(); c.ws=nf(); };
  auto readTL = [&](TLp& t) { t.ln1=nf(); t.ln2=nf(); t.w1=nf(); t.w2=nf();
    t.wo=nf(); t.wqkv=nf(); };
  CBp dec[2], enc[2]; TLp tdec[3], tenc[3];
  readCB(dec[0]); readCB(dec[1]);
  const float* embed = nf();
  readCB(enc[0]); readCB(enc[1]);
  const float* enc_ln_b = nf(); const float* enc_ln_g = nf();
  const float* out_b = nf();    const float* out_w = nf();
  const float* pos = nf();
  readTL(tdec[0]); readTL(tdec[1]); readTL(tdec[2]);
  readTL(tenc[0]); readTL(tenc[1]); readTL(tenc[2]);
  const float* t_enc_ln_b = nf(); const float* t_enc_ln_g = nf();
  const float* codebook = nf();
  const int* tokens = (const int*)d_in[p++];

  // ---- workspace bump allocator -------------------------------------------
  size_t off = 0;
  auto alloc_h = [&](size_t n) -> half_t* {
    half_t* q = (half_t*)((char*)d_ws + off);
    off = (off + n * sizeof(half_t) + 255) & ~(size_t)255; return q; };
  auto alloc_f = [&](size_t n) -> float* {
    float* q = (float*)((char*)d_ws + off);
    off = (off + n * sizeof(float) + 255) & ~(size_t)255; return q; };
  auto alloc_i = [&](size_t n) -> int* {
    int* q = (int*)((char*)d_ws + off);
    off = (off + n * sizeof(int) + 255) & ~(size_t)255; return q; };

  struct ConvW { half_t *w1h, *w2h, *wsh;
                 float *a1, *bt1, *a2, *bt2, *as, *bts; };
  auto alloc_cw = [&](ConvW& cw) {
    cw.w1h = alloc_h(9 * 256 * 256); cw.w2h = alloc_h(9 * 256 * 256);
    cw.wsh = alloc_h(256 * 256);
    cw.a1 = alloc_f(256); cw.bt1 = alloc_f(256);
    cw.a2 = alloc_f(256); cw.bt2 = alloc_f(256);
    cw.as = alloc_f(256); cw.bts = alloc_f(256); };
  ConvW encW[2], decW[2];
  alloc_cw(encW[0]); alloc_cw(encW[1]); alloc_cw(decW[0]); alloc_cw(decW[1]);

  struct TLW { half_t *wqkv, *wo, *w1, *w2; };
  auto alloc_tw = [&](TLW& t) {
    t.wqkv = alloc_h(768 * 256); t.wo = alloc_h(256 * 256);
    t.w1 = alloc_h(1024 * 256);  t.w2 = alloc_h(256 * 1024); };
  TLW tencW[3], tdecW[3];
  for (int i = 0; i < 3; ++i) alloc_tw(tencW[i]);
  for (int i = 0; i < 3; ++i) alloc_tw(tdecW[i]);

  half_t* cbh = alloc_h((size_t)KCODE * 256);
  float*  cbn = alloc_f(KCODE);
  half_t* owh = alloc_h(16 * 256);
  half_t* zpad = alloc_h(256);          // zero page for OOB conv taps

  half_t* xe  = alloc_h((size_t)64 * 32 * 32 * 256);
  half_t* r16 = alloc_h((size_t)64 * 16 * 16 * 256);
  half_t* h16 = alloc_h((size_t)64 * 16 * 16 * 256);
  half_t* x16 = alloc_h((size_t)64 * 16 * 16 * 256);
  half_t* r8  = alloc_h((size_t)64 * 8 * 8 * 256);
  half_t* h8  = alloc_h((size_t)64 * 8 * 8 * 256);
  half_t* x8  = alloc_h((size_t)64 * 8 * 8 * 256);
  half_t* zA  = alloc_h((size_t)NTOK * 256);
  half_t* zB  = alloc_h((size_t)NTOK * 256);
  half_t* u   = alloc_h((size_t)NTOK * 256);
  half_t* ob  = alloc_h((size_t)NTOK * 256);
  half_t* qkvb= alloc_h((size_t)NTOK * 768);
  half_t* ffb = alloc_h((size_t)NTOK * 1024);
  float*  Dsc = alloc_f((size_t)NTOK * KCODE);
  int*    idx = alloc_i(NTOK);
  half_t* h32 = alloc_h((size_t)64 * 32 * 32 * 256);
  half_t* y32 = alloc_h((size_t)64 * 32 * 32 * 256);
  half_t* r32 = xe;   // xe dead after encoder block 1 -> reuse for dec residual

  // ---- launch helpers ------------------------------------------------------
  auto ew = [&](int n) { return dim3((n + 255) / 256); };
  auto gemm = [&](const half_t* A, int lda, const half_t* Bw, int K,
                  half_t* Yh, float* Yf, int ldy, const float* bias,
                  const half_t* resid, int M, int N, int relu) {
    if ((N & 63) == 0) {
      dim3 g(M / 32, N / 64);
      gemm_wmma<2, 4><<<g, 32, 0, stream>>>(A, lda, Bw, K, Yh, Yf, ldy, bias,
                                            resid, relu);
    } else {
      dim3 g(M / 32, N / 16);
      gemm_wmma<2, 1><<<g, 32, 0, stream>>>(A, lda, Bw, K, Yh, Yf, ldy, bias,
                                            resid, relu);
    }
  };
  auto conv = [&](const half_t* X, const half_t* Wk, const float* al,
                  const float* be, const half_t* resid, half_t* Y,
                  int IH, int OH, int KH, int stride, int pad, int relu,
                  int transposed) {
    dim3 g((64 * OH * OH) / 16, 256 / 64);
    const int lw = (OH == 8) ? 3 : ((OH == 16) ? 4 : 5);
    if (transposed) {
      if (KH == 3)
        convlike_wmma<3,1><<<g, 32, 0, stream>>>(X, Wk, al, be, resid, Y, zpad,
                                                 IH, lw, stride, pad, relu);
      else
        convlike_wmma<1,1><<<g, 32, 0, stream>>>(X, Wk, al, be, resid, Y, zpad,
                                                 IH, lw, stride, pad, relu);
    } else {
      if (KH == 3)
        convlike_wmma<3,0><<<g, 32, 0, stream>>>(X, Wk, al, be, resid, Y, zpad,
                                                 IH, lw, stride, pad, relu);
      else
        convlike_wmma<1,0><<<g, 32, 0, stream>>>(X, Wk, al, be, resid, Y, zpad,
                                                 IH, lw, stride, pad, relu);
    }
  };

  // ---- prep: weight repacks + folded BN epilogues --------------------------
  zero_h_kernel<<<1, 256, 0, stream>>>(zpad, 256);
  const int c9 = 9 * 256 * 256, c1 = 256 * 256;
  for (int i = 0; i < 2; ++i) {
    repack_conv_kernel<<<ew(c9), 256, 0, stream>>>(enc[i].w1, encW[i].w1h, 256, 256, 3, 0, c9);
    repack_conv_kernel<<<ew(c9), 256, 0, stream>>>(enc[i].w2, encW[i].w2h, 256, 256, 3, 0, c9);
    repack_conv_kernel<<<ew(c1), 256, 0, stream>>>(enc[i].ws, encW[i].wsh, 256, 256, 1, 0, c1);
    repack_conv_kernel<<<ew(c9), 256, 0, stream>>>(dec[i].w1, decW[i].w1h, 256, 256, 3, 1, c9);
    repack_conv_kernel<<<ew(c9), 256, 0, stream>>>(dec[i].w2, decW[i].w2h, 256, 256, 3, 0, c9);
    repack_conv_kernel<<<ew(c1), 256, 0, stream>>>(dec[i].ws, decW[i].wsh, 256, 256, 1, 1, c1);
    alphabeta_kernel<<<1, 256, 0, stream>>>(enc[i].b1, enc[i].g1, enc[i].be1, encW[i].a1, encW[i].bt1, 256);
    alphabeta_kernel<<<1, 256, 0, stream>>>(enc[i].b2, enc[i].g2, enc[i].be2, encW[i].a2, encW[i].bt2, 256);
    alphabeta_kernel<<<1, 256, 0, stream>>>(enc[i].bs, enc[i].gs, enc[i].bes, encW[i].as, encW[i].bts, 256);
    alphabeta_kernel<<<1, 256, 0, stream>>>(dec[i].b1, dec[i].g1, dec[i].be1, decW[i].a1, decW[i].bt1, 256);
    alphabeta_kernel<<<1, 256, 0, stream>>>(dec[i].b2, dec[i].g2, dec[i].be2, decW[i].a2, decW[i].bt2, 256);
    alphabeta_kernel<<<1, 256, 0, stream>>>(dec[i].bs, dec[i].gs, dec[i].bes, decW[i].as, decW[i].bts, 256);
  }
  for (int i = 0; i < 3; ++i) {
    cvt_h_kernel<<<ew(768 * 256), 256, 0, stream>>>(tenc[i].wqkv, tencW[i].wqkv, 768 * 256);
    cvt_h_kernel<<<ew(256 * 256), 256, 0, stream>>>(tenc[i].wo,   tencW[i].wo,   256 * 256);
    cvt_h_kernel<<<ew(1024 * 256), 256, 0, stream>>>(tenc[i].w1,  tencW[i].w1,  1024 * 256);
    cvt_h_kernel<<<ew(1024 * 256), 256, 0, stream>>>(tenc[i].w2,  tencW[i].w2,  1024 * 256);
    cvt_h_kernel<<<ew(768 * 256), 256, 0, stream>>>(tdec[i].wqkv, tdecW[i].wqkv, 768 * 256);
    cvt_h_kernel<<<ew(256 * 256), 256, 0, stream>>>(tdec[i].wo,   tdecW[i].wo,   256 * 256);
    cvt_h_kernel<<<ew(1024 * 256), 256, 0, stream>>>(tdec[i].w1,  tdecW[i].w1,  1024 * 256);
    cvt_h_kernel<<<ew(1024 * 256), 256, 0, stream>>>(tdec[i].w2,  tdecW[i].w2,  1024 * 256);
  }
  cvt_h_kernel<<<ew(KCODE * 256), 256, 0, stream>>>(codebook, cbh, KCODE * 256);
  rownorm_kernel<<<KCODE, 32, 0, stream>>>(codebook, cbn);
  cvt_h_kernel<<<ew(16 * 256), 256, 0, stream>>>(out_w, owh, 16 * 256);

  // ---- forward -------------------------------------------------------------
  embed_kernel<<<ew(64 * 32 * 32 * 256), 256, 0, stream>>>(tokens, embed, xe,
                                                           64 * 32 * 32 * 256);
  // encoder conv blocks: 32 -> 16 -> 8
  conv(xe, encW[0].wsh, encW[0].as, encW[0].bts, nullptr, r16, 32, 16, 1, 2, 0, 0, 0);
  conv(xe, encW[0].w1h, encW[0].a1, encW[0].bt1, nullptr, h16, 32, 16, 3, 2, 1, 1, 0);
  conv(h16, encW[0].w2h, encW[0].a2, encW[0].bt2, r16, x16, 16, 16, 3, 1, 1, 1, 0);
  conv(x16, encW[1].wsh, encW[1].as, encW[1].bts, nullptr, r8, 16, 8, 1, 2, 0, 0, 0);
  conv(x16, encW[1].w1h, encW[1].a1, encW[1].bt1, nullptr, h8, 16, 8, 3, 2, 1, 1, 0);
  conv(h8, encW[1].w2h, encW[1].a2, encW[1].bt2, r8, x8, 8, 8, 3, 1, 1, 1, 0);
  // channel-last LN + positional embedding -> z
  ln_rows<<<NTOK, 32, 0, stream>>>(x8, enc_ln_g, enc_ln_b, pos, zA);
  // transformer layers (shared routine)
  auto tlayer = [&](const TLp& P, const TLW& W) {
    ln_rows<<<NTOK, 32, 0, stream>>>(zA, P.ln1, nullptr, nullptr, u);
    gemm(u, 256, W.wqkv, 256, qkvb, nullptr, 768, nullptr, nullptr, NTOK, 768, 0);
    attention_wmma<<<256, 128, 0, stream>>>(qkvb, ob);
    gemm(ob, 256, W.wo, 256, zB, nullptr, 256, nullptr, zA, NTOK, 256, 0);
    ln_rows<<<NTOK, 32, 0, stream>>>(zB, P.ln2, nullptr, nullptr, u);
    gemm(u, 256, W.w1, 256, ffb, nullptr, 1024, nullptr, nullptr, NTOK, 1024, 1);
    gemm(ffb, 1024, W.w2, 1024, zA, nullptr, 256, nullptr, zB, NTOK, 256, 0);
  };
  for (int i = 0; i < 3; ++i) tlayer(tenc[i], tencW[i]);
  // final encoder LN -> u, then VQ
  ln_rows<<<NTOK, 32, 0, stream>>>(zA, t_enc_ln_g, t_enc_ln_b, nullptr, u);
  gemm(u, 256, cbh, 256, nullptr, Dsc, KCODE, nullptr, nullptr, NTOK, KCODE, 0);
  vq_argmin_kernel<<<NTOK, 256, 0, stream>>>(Dsc, cbn, idx);
  vq_gather_kernel<<<ew(NTOK * 256), 256, 0, stream>>>(cbh, idx, zA, NTOK * 256);
  // decoder transformer layers
  for (int i = 0; i < 3; ++i) tlayer(tdec[i], tdecW[i]);
  // decoder conv blocks: 8 -> 16 -> 32 (transposed convs via gather)
  conv(zA, decW[0].wsh, decW[0].as, decW[0].bts, nullptr, r16, 8, 16, 1, 2, 0, 0, 1);
  conv(zA, decW[0].w1h, decW[0].a1, decW[0].bt1, nullptr, h16, 8, 16, 3, 2, 1, 1, 1);
  conv(h16, decW[0].w2h, decW[0].a2, decW[0].bt2, r16, x16, 16, 16, 3, 1, 1, 1, 0);
  conv(x16, decW[1].wsh, decW[1].as, decW[1].bts, nullptr, r32, 16, 32, 1, 2, 0, 0, 1);
  conv(x16, decW[1].w1h, decW[1].a1, decW[1].bt1, nullptr, h32, 16, 32, 3, 2, 1, 1, 1);
  conv(h32, decW[1].w2h, decW[1].a2, decW[1].bt2, r32, y32, 32, 32, 3, 1, 1, 1, 0);
  // output projection -> fp32 logits
  gemm(y32, 256, owh, 256, nullptr, (float*)d_out, 16, out_b, nullptr,
       64 * 32 * 32, 16, 0);
}
